// Informer_78030965833966
// MI455X (gfx1250) — matrix operand order repaired and checked by reading, not compiled, not run on hardware
//
#include <hip/hip_runtime.h>
#include <cstdint>

#define DEVFN __device__ __forceinline__

typedef __attribute__((ext_vector_type(16))) __bf16 v16bf;
typedef __attribute__((ext_vector_type(8)))  float  v8f;
typedef __attribute__((ext_vector_type(4)))  unsigned int u32x4;
typedef __attribute__((ext_vector_type(8)))  int iv8;
typedef __attribute__((ext_vector_type(4)))  int iv4;

union FragU { v16bf v; u32x4 q[2]; };

// Problem constants
constexpr int Bc   = 4;
constexpr int Lc   = 1024;
constexpr int EINc = 64;
constexpr int DMc  = 512;
constexpr int Hc   = 8;
constexpr int HDc  = 64;
constexpr int DFFc = 2048;
constexpr int NLc  = 3;
constexpr int COUTc= 64;
constexpr int M4c  = Bc * Lc;   // 4096 rows
constexpr int BHc  = Bc * Hc;   // 32 attention batches

DEVFN unsigned short f2bf(float x) {
  unsigned int u = __float_as_uint(x);
  u += 0x7fffu + ((u >> 16) & 1u);           // round-to-nearest-even
  return (unsigned short)(u >> 16);
}

DEVFN float pos_enc(int pos, int n) {
  // pe[:,2i]=sin(pos*exp(-2i*ln(1e4)/512)), pe[:,2i+1]=cos(...)
  float f = __expf((float)(n & ~1) * (-9.210340371976184f / 512.0f));
  float a = (float)pos * f;
  return (n & 1) ? __cosf(a) : __sinf(a);
}

// ---------------------------------------------------------------------------
// TDM tile load: DMA a [tileRows=128] x [32 halves] tile of a row-major bf16
// matrix (row pitch K elements) into LDS with 32B padding after each 64B row
// (-> 96B LDS row pitch). Remaining-extent tensor dims let TDM zero-fill OOB.
// Descriptor packing per CDNA5 ISA 8.3/8.4 (group0 128b, group1 256b).
// This toolchain exposes the 6-arg builtin:
//   (uint32x4 g0, int32x8 g1, int32x4 g2, int32x4 g3, int32x8 g4, int cpol)
// ---------------------------------------------------------------------------
DEVFN void tdm_tile_load(unsigned lds_addr, const unsigned short* gbase,
                         int rowStart, int k0, int K, int totRows) {
  unsigned long long ga = (unsigned long long)(uintptr_t)gbase +
      ((unsigned long long)(unsigned)rowStart * (unsigned long long)(unsigned)K +
       (unsigned long long)(unsigned)k0) * 2ull;
  unsigned d0 = (unsigned)(K - k0);        // remaining cols (elements)
  unsigned d1 = (unsigned)(totRows - rowStart); // remaining rows
  u32x4 g0;
  g0[0] = 1u;                                   // count=1, is_restore=0
  g0[1] = lds_addr;                             // LDS byte address
  g0[2] = (unsigned)ga;                         // global_addr[95:64]
  g0[3] = (unsigned)(ga >> 32) | 0x80000000u;   // global_addr hi | type=2
  iv8 g1;
  // data_size=2B(1), pad_enable, pad_interval=16 DW (64B), pad_amount=8 DW (32B)
  g1[0] = (1 << 16) | (1 << 20) | (3 << 22) | (7 << 25);
  g1[1] = (int)((d0 & 0xFFFFu) << 16);                    // tensor_dim0 lo16
  g1[2] = (int)((d0 >> 16) | ((d1 & 0xFFFFu) << 16));     // dim0 hi | dim1 lo
  g1[3] = (int)((d1 >> 16) | (32u << 16));                // dim1 hi | tile_dim0=32
  g1[4] = 128;                                            // tile_dim1=128, tile_dim2=0
  g1[5] = K;                                              // tensor_dim0_stride lo32
  g1[6] = 0;
  g1[7] = 0;
  iv4 z4 = {0, 0, 0, 0};
  iv8 z8 = {0, 0, 0, 0, 0, 0, 0, 0};
  __builtin_amdgcn_tensor_load_to_lds(g0, g1, z4, z4, z8, 0);
}

enum {
  EPI_BIAS = 0,      // outF[m*N+n] = acc + bias[n]
  EPI_BIAS_RES = 1,  // outF[m*N+n] = acc + bias[n] + res[m*N+n]
  EPI_EMBED = 2,     // outF & outH = acc + bias[n] + PE(m%L, n)
  EPI_QKV = 3,       // outH bf16 at [b,h,l,d] layout, +bias
  EPI_VT = 4,        // outH bf16 at [b,h,d,l] layout (transposed V), +bias
  EPI_SCORE = 5,     // outF[z*strideO + m*N + n] = acc (raw QK)
  EPI_GELU = 6,      // outH bf16 = gelu(acc + bias[n])
  EPI_ATTN = 7       // outH bf16 at [b, u, h*64+d] concat layout
};

// ---------------------------------------------------------------------------
// Tiled bf16 WMMA GEMM: C[M,N] = A[M,K] * Bt[N,K]^T, fp32 accumulate.
// Block: 256 threads (8 waves), tile 128x128, K-step 32, TDM double-buffered
// LDS staging. Wave grid 4(M) x 2(N); each wave: 2x4 tiles of 16x16 via
// v_wmma_f32_16x16x32_bf16.
// ---------------------------------------------------------------------------
template <int EPI>
__global__ __launch_bounds__(256)
void gemm_bf16(const unsigned short* __restrict__ Ag,
               const unsigned short* __restrict__ Bg,
               const float* __restrict__ bias,
               const float* __restrict__ res,
               float* __restrict__ outF,
               unsigned short* __restrict__ outH,
               int M, int N, int K,
               long long strideA, long long strideB, long long strideO) {
  // 2 buffers x 128 rows x (32 data + 16 pad) halves  (96B row pitch)
  __shared__ unsigned short lsA[2][128 * 48];
  __shared__ unsigned short lsB[2][128 * 48];
  constexpr unsigned TILE_B = 128u * 48u * 2u;

  const int tid  = threadIdx.x;
  const int lane = tid & 31;
  const int wave = tid >> 5;
  const int wm   = wave & 3;        // 0..3 along M
  const int wn   = wave >> 2;       // 0..1 along N
  const int blockN = blockIdx.x * 128;
  const int blockM = blockIdx.y * 128;
  const int z      = blockIdx.z;

  const unsigned short* A  = Ag + (size_t)z * (size_t)strideA;
  const unsigned short* Bt = Bg + (size_t)z * (size_t)strideB;

  const unsigned ldsA0 = (unsigned)(uintptr_t)(&lsA[0][0]);
  const unsigned ldsB0 = (unsigned)(uintptr_t)(&lsB[0][0]);

  v8f acc[2][4];
#pragma unroll
  for (int i = 0; i < 2; i++)
#pragma unroll
    for (int j = 0; j < 4; j++)
      acc[i][j] = v8f{0.f, 0.f, 0.f, 0.f, 0.f, 0.f, 0.f, 0.f};

  // prologue: DMA first K-tile, wait, release block
  if (wave == 0) {
    tdm_tile_load(ldsA0, A, blockM, 0, K, M);
    tdm_tile_load(ldsB0, Bt, blockN, 0, K, N);
    __builtin_amdgcn_s_wait_tensorcnt(0);
  }
  __syncthreads();

  int buf = 0;
  for (int k0 = 0; k0 < K; k0 += 32) {
    const bool more = (k0 + 32 < K);
    // overlap: DMA next tile into the other buffer while computing this one
    if (more && wave == 0) {
      tdm_tile_load(ldsA0 + (unsigned)(buf ^ 1) * TILE_B, A, blockM, k0 + 32, K, M);
      tdm_tile_load(ldsB0 + (unsigned)(buf ^ 1) * TILE_B, Bt, blockN, k0 + 32, K, N);
    }

    // ---- load fragments (16-bit A 16x32 layout: lanes 0-15 K 0..7|16..23,
    //      lanes 16-31 K 8..15|24..31; B uses the same K order) ----
    FragU a[2], b[4];
    const unsigned short* baseA = &lsA[buf][0];
    const unsigned short* baseB = &lsB[buf][0];
    const int ac = (lane < 16) ? 0 : 8;
#pragma unroll
    for (int i = 0; i < 2; i++) {
      const unsigned short* p = baseA + (wm * 32 + i * 16 + (lane & 15)) * 48 + ac;
      a[i].q[0] = *reinterpret_cast<const u32x4*>(p);
      a[i].q[1] = *reinterpret_cast<const u32x4*>(p + 16);
    }
    const int bc = (lane < 16) ? 0 : 8;
#pragma unroll
    for (int j = 0; j < 4; j++) {
      const unsigned short* p = baseB + (wn * 64 + j * 16 + (lane & 15)) * 48 + bc;
      b[j].q[0] = *reinterpret_cast<const u32x4*>(p);
      b[j].q[1] = *reinterpret_cast<const u32x4*>(p + 16);
    }

#pragma unroll
    for (int i = 0; i < 2; i++)
#pragma unroll
      for (int j = 0; j < 4; j++)
        acc[i][j] = __builtin_amdgcn_wmma_f32_16x16x32_bf16(
            false, a[i].v, false, b[j].v, (short)0, acc[i][j], false, false);

    if (more) {
      if (wave == 0) __builtin_amdgcn_s_wait_tensorcnt(0);
      __syncthreads();
      buf ^= 1;
    }
  }

  // ---- epilogue: C fragment layout — VGPR r, lane l: M=r+(l<16?0:8), N=l%16
#pragma unroll
  for (int i = 0; i < 2; i++) {
#pragma unroll
    for (int j = 0; j < 4; j++) {
      int mBase = blockM + wm * 32 + i * 16 + ((lane & 16) ? 8 : 0);
      int n     = blockN + wn * 64 + j * 16 + (lane & 15);
#pragma unroll
      for (int r = 0; r < 8; r++) {
        int m = mBase + r;
        if (m >= M || n >= N) continue;
        float val = acc[i][j][r];
        if constexpr (EPI == EPI_BIAS) {
          outF[(size_t)m * N + n] = val + bias[n];
        } else if constexpr (EPI == EPI_BIAS_RES) {
          outF[(size_t)m * N + n] = val + bias[n] + res[(size_t)m * N + n];
        } else if constexpr (EPI == EPI_EMBED) {
          float o = val + bias[n] + pos_enc(m & (Lc - 1), n);
          outF[(size_t)m * N + n] = o;
          outH[(size_t)m * N + n] = f2bf(o);
        } else if constexpr (EPI == EPI_QKV) {
          int bb = m >> 10, l = m & (Lc - 1);
          int h = n >> 6, d = n & 63;
          outH[((((size_t)bb * Hc + h) << 10) + l) * HDc + d] = f2bf(val + bias[n]);
        } else if constexpr (EPI == EPI_VT) {
          int bb = m >> 10, l = m & (Lc - 1);
          int h = n >> 6, d = n & 63;
          outH[(((size_t)bb * Hc + h) * HDc + d) * Lc + l] = f2bf(val + bias[n]);
        } else if constexpr (EPI == EPI_SCORE) {
          outF[(size_t)z * (size_t)strideO + (size_t)m * N + n] = val;
        } else if constexpr (EPI == EPI_GELU) {
          float x = val + bias[n];
          float g = 0.5f * x * (1.0f + erff(x * 0.7071067811865475f));
          outH[(size_t)m * N + n] = f2bf(g);
        } else if constexpr (EPI == EPI_ATTN) {
          int bb = z >> 3, h = z & 7;
          outH[(((size_t)bb * Lc + m) * DMc) + (size_t)h * HDc + n] = f2bf(val);
        }
      }
    }
  }
}

// ---------------------------------------------------------------------------
// fp32 -> bf16 convert; fp32 [K,N] -> bf16 [N,K] transpose-convert
// ---------------------------------------------------------------------------
__global__ void conv_bf16(const float* __restrict__ s, unsigned short* __restrict__ d, int n) {
  int t = blockIdx.x * 256 + threadIdx.x;
  if (t < n) d[t] = f2bf(s[t]);
}

__global__ void tconv_bf16(const float* __restrict__ s, unsigned short* __restrict__ d, int K, int N) {
  int t = blockIdx.x * 256 + threadIdx.x;
  if (t < K * N) {
    int k = t / N, n = t - k * N;
    d[(size_t)n * K + k] = f2bf(s[t]);
  }
}

// ---------------------------------------------------------------------------
// ProbSparse measure: M[z,q] = max_j S[z,q,idx[q,j]] - mean_j S[z,q,idx[q,j]]
// grid (L, BH), block 256
// ---------------------------------------------------------------------------
__global__ void prob_measure(const float* __restrict__ S, const int* __restrict__ idx,
                             float* __restrict__ Mout) {
  int t = threadIdx.x, q = blockIdx.x, z = blockIdx.y;
  const float* row = S + ((size_t)z * Lc + q) * Lc;
  const int* ids = idx + (size_t)q * Lc;
  float mx = -3.402823466e38f, sm = 0.f;
  for (int j = t; j < Lc; j += 256) {
    float v = row[ids[j]];
    mx = fmaxf(mx, v);
    sm += v;
  }
  __shared__ float r1[256], r2[256];
  r1[t] = mx; r2[t] = sm;
  __syncthreads();
  for (int s = 128; s > 0; s >>= 1) {
    if (t < s) { r1[t] = fmaxf(r1[t], r1[t + s]); r2[t] += r2[t + s]; }
    __syncthreads();
  }
  if (t == 0) Mout[(size_t)z * Lc + q] = r1[0] - r2[0] * (1.0f / (float)Lc);
}

// ---------------------------------------------------------------------------
// Full descending sort (== top_k with k=L): bitonic, 1024 threads per (b,h)
// tie-break: lower index first (matches jax.lax.top_k)
// ---------------------------------------------------------------------------
__global__ void topk_sort(const float* __restrict__ Mbuf, int* __restrict__ top) {
  __shared__ float sk[1024];
  __shared__ int sv[1024];
  int t = threadIdx.x, z = blockIdx.x;
  sk[t] = Mbuf[(size_t)z * Lc + t];
  sv[t] = t;
  __syncthreads();
  for (int size = 2; size <= 1024; size <<= 1) {
    for (int stride = size >> 1; stride > 0; stride >>= 1) {
      int p = t ^ stride;
      if (p > t) {
        float ka = sk[t], kb = sk[p];
        int va = sv[t], vb = sv[p];
        bool desc = ((t & size) == 0);
        bool aFirst = (ka > kb) || (ka == kb && va < vb);
        if (desc ? !aFirst : aFirst) {
          sk[t] = kb; sk[p] = ka; sv[t] = vb; sv[p] = va;
        }
      }
      __syncthreads();
    }
  }
  top[(size_t)z * Lc + t] = sv[t];
}

// ---------------------------------------------------------------------------
// Softmax of permuted score rows -> bf16 probabilities
// P[z,u,:] = softmax(S[z,top[z,u],:] * SCALE)   grid (L, BH), block 256
// ---------------------------------------------------------------------------
__global__ void softmax_rows(const float* __restrict__ S, const int* __restrict__ top,
                             unsigned short* __restrict__ P) {
  int t = threadIdx.x, u = blockIdx.x, z = blockIdx.y;
  int row = top[(size_t)z * Lc + u];
  const float* s = S + ((size_t)z * Lc + row) * Lc;
  unsigned short* p = P + ((size_t)z * Lc + u) * Lc;
  float v[4];
#pragma unroll
  for (int c = 0; c < 4; c++) v[c] = s[t + 256 * c] * 0.125f;  // SCALE=1/sqrt(64)
  float mx = fmaxf(fmaxf(v[0], v[1]), fmaxf(v[2], v[3]));
  __shared__ float red[256];
  red[t] = mx;
  __syncthreads();
  for (int st = 128; st > 0; st >>= 1) {
    if (t < st) red[t] = fmaxf(red[t], red[t + st]);
    __syncthreads();
  }
  mx = red[0];
  __syncthreads();
  float e[4], sm = 0.f;
#pragma unroll
  for (int c = 0; c < 4; c++) { e[c] = __expf(v[c] - mx); sm += e[c]; }
  red[t] = sm;
  __syncthreads();
  for (int st = 128; st > 0; st >>= 1) {
    if (t < st) red[t] += red[t + st];
    __syncthreads();
  }
  float inv = 1.0f / red[0];
#pragma unroll
  for (int c = 0; c < 4; c++) p[t + 256 * c] = f2bf(e[c] * inv);
}

// ---------------------------------------------------------------------------
// LayerNorm over DM=512 per row; writes fp32 + bf16 copies. In-place safe.
// grid M4, block 256
// ---------------------------------------------------------------------------
__global__ void ln_kernel(const float* __restrict__ src, const float* __restrict__ g,
                          const float* __restrict__ b, float* __restrict__ xf,
                          unsigned short* __restrict__ xb) {
  int t = threadIdx.x;
  size_t base = (size_t)blockIdx.x * DMc;
  float v0 = src[base + t], v1 = src[base + t + 256];
  __shared__ float red[256];
  red[t] = v0 + v1;
  __syncthreads();
  for (int s = 128; s > 0; s >>= 1) {
    if (t < s) red[t] += red[t + s];
    __syncthreads();
  }
  float mean = red[0] * (1.0f / (float)DMc);
  __syncthreads();
  float d0 = v0 - mean, d1 = v1 - mean;
  red[t] = d0 * d0 + d1 * d1;
  __syncthreads();
  for (int s = 128; s > 0; s >>= 1) {
    if (t < s) red[t] += red[t + s];
    __syncthreads();
  }
  float inv = rsqrtf(red[0] * (1.0f / (float)DMc) + 1e-5f);
  float o0 = d0 * inv * g[t] + b[t];
  float o1 = d1 * inv * g[t + 256] + b[t + 256];
  xf[base + t] = o0;
  xf[base + t + 256] = o1;
  xb[base + t] = f2bf(o0);
  xb[base + t + 256] = f2bf(o1);
}

// ---------------------------------------------------------------------------
extern "C" void kernel_launch(void* const* d_in, const int* in_sizes, int n_in,
                              void* d_out, int out_size, void* d_ws, size_t ws_size,
                              hipStream_t stream) {
  (void)in_sizes; (void)n_in; (void)out_size; (void)ws_size;
  const float* x_enc      = (const float*)d_in[0];
  const int*   sample_idx = (const int*)d_in[1];
  const float* W_emb = (const float*)d_in[2];  const float* b_emb = (const float*)d_in[3];
  const float* Wq = (const float*)d_in[4];     const float* bq = (const float*)d_in[5];
  const float* Wk = (const float*)d_in[6];     const float* bk = (const float*)d_in[7];
  const float* Wv = (const float*)d_in[8];     const float* bv = (const float*)d_in[9];
  const float* Wo = (const float*)d_in[10];    const float* bo = (const float*)d_in[11];
  const float* W1 = (const float*)d_in[12];    const float* b1 = (const float*)d_in[13];
  const float* W2 = (const float*)d_in[14];    const float* b2 = (const float*)d_in[15];
  const float* g1 = (const float*)d_in[16];    const float* be1 = (const float*)d_in[17];
  const float* g2 = (const float*)d_in[18];    const float* be2 = (const float*)d_in[19];
  const float* gN = (const float*)d_in[20];    const float* bN = (const float*)d_in[21];
  const float* W_proj = (const float*)d_in[22];const float* b_proj = (const float*)d_in[23];
  float* out = (float*)d_out;

  // ---- workspace carving ----
  char* wsb = (char*)d_ws;
  size_t off = 0;
  auto ALLOC = [&](size_t bytes) -> char* {
    char* p = wsb + off;
    off += (bytes + 255) & ~(size_t)255;
    return p;
  };
  float* xf  = (float*)ALLOC((size_t)M4c * DMc * 4);
  float* zf  = (float*)ALLOC((size_t)M4c * DMc * 4);
  unsigned short* xb    = (unsigned short*)ALLOC((size_t)M4c * DMc * 2);
  unsigned short* attnB = (unsigned short*)ALLOC((size_t)M4c * DMc * 2);
  unsigned short* hidB  = (unsigned short*)ALLOC((size_t)M4c * DFFc * 2);
  unsigned short* qB    = (unsigned short*)ALLOC((size_t)BHc * Lc * HDc * 2);
  unsigned short* kB    = (unsigned short*)ALLOC((size_t)BHc * Lc * HDc * 2);
  unsigned short* vtB   = (unsigned short*)ALLOC((size_t)BHc * HDc * Lc * 2);
  float* S = (float*)ALLOC((size_t)BHc * Lc * Lc * 4);           // 128 MB
  unsigned short* P = (unsigned short*)ALLOC((size_t)BHc * Lc * Lc * 2);  // 64 MB
  float* Mb = (float*)ALLOC((size_t)BHc * Lc * 4);
  int* top  = (int*)ALLOC((size_t)BHc * Lc * 4);
  unsigned short* xeB   = (unsigned short*)ALLOC((size_t)M4c * EINc * 2);
  unsigned short* WembT = (unsigned short*)ALLOC((size_t)DMc * EINc * 2);
  unsigned short* WqT = (unsigned short*)ALLOC((size_t)NLc * DMc * DMc * 2);
  unsigned short* WkT = (unsigned short*)ALLOC((size_t)NLc * DMc * DMc * 2);
  unsigned short* WvT = (unsigned short*)ALLOC((size_t)NLc * DMc * DMc * 2);
  unsigned short* WoT = (unsigned short*)ALLOC((size_t)NLc * DMc * DMc * 2);
  unsigned short* W1T = (unsigned short*)ALLOC((size_t)NLc * DMc * DFFc * 2);
  unsigned short* W2T = (unsigned short*)ALLOC((size_t)NLc * DFFc * DMc * 2);
  unsigned short* WpT = (unsigned short*)ALLOC((size_t)DMc * COUTc * 2);

  dim3 blk(256);
  auto cdiv = [](int a, int b) { return (a + b - 1) / b; };

  // ---- weight/input conversion (deterministic every call) ----
  conv_bf16<<<cdiv(M4c * EINc, 256), blk, 0, stream>>>(x_enc, xeB, M4c * EINc);
  tconv_bf16<<<cdiv(EINc * DMc, 256), blk, 0, stream>>>(W_emb, WembT, EINc, DMc);
  for (int l = 0; l < NLc; l++) {
    size_t o = (size_t)l * DMc * DMc;
    tconv_bf16<<<cdiv(DMc * DMc, 256), blk, 0, stream>>>(Wq + o, WqT + o, DMc, DMc);
    tconv_bf16<<<cdiv(DMc * DMc, 256), blk, 0, stream>>>(Wk + o, WkT + o, DMc, DMc);
    tconv_bf16<<<cdiv(DMc * DMc, 256), blk, 0, stream>>>(Wv + o, WvT + o, DMc, DMc);
    tconv_bf16<<<cdiv(DMc * DMc, 256), blk, 0, stream>>>(Wo + o, WoT + o, DMc, DMc);
    size_t of = (size_t)l * DMc * DFFc;
    tconv_bf16<<<cdiv(DMc * DFFc, 256), blk, 0, stream>>>(W1 + of, W1T + of, DMc, DFFc);
    tconv_bf16<<<cdiv(DFFc * DMc, 256), blk, 0, stream>>>(W2 + of, W2T + of, DFFc, DMc);
  }
  tconv_bf16<<<cdiv(DMc * COUTc, 256), blk, 0, stream>>>(W_proj, WpT, DMc, COUTc);

#define GEMM(EPI, A_, B_, bias_, res_, oF_, oH_, Mm, Nn, Kk, batch, sA, sB, sO)      \
  gemm_bf16<EPI><<<dim3(((Nn) + 127) / 128, ((Mm) + 127) / 128, (batch)), blk, 0,    \
                   stream>>>(A_, B_, bias_, res_, oF_, oH_, Mm, Nn, Kk,              \
                             (long long)(sA), (long long)(sB), (long long)(sO))

  // ---- embedding: x = x_enc @ W_emb + b_emb + PE ----
  GEMM(EPI_EMBED, xeB, WembT, b_emb, nullptr, xf, xb, M4c, DMc, EINc, 1, 0, 0, 0);

  for (int l = 0; l < NLc; l++) {
    const size_t ow = (size_t)l * DMc * DMc;
    // Q,K (head layout), V (transposed head layout)
    GEMM(EPI_QKV, xb, WqT + ow, bq + l * DMc, nullptr, nullptr, qB, M4c, DMc, DMc, 1, 0, 0, 0);
    GEMM(EPI_QKV, xb, WkT + ow, bk + l * DMc, nullptr, nullptr, kB, M4c, DMc, DMc, 1, 0, 0, 0);
    GEMM(EPI_VT,  xb, WvT + ow, bv + l * DMc, nullptr, nullptr, vtB, M4c, DMc, DMc, 1, 0, 0, 0);
    // S[z] = q[z] @ k[z]^T  (batched over b*h = 32)
    GEMM(EPI_SCORE, qB, kB, nullptr, nullptr, S, nullptr, Lc, Lc, HDc, BHc,
         (long long)Lc * HDc, (long long)Lc * HDc, (long long)Lc * Lc);
    // M = max - mean over sampled columns, full sort, softmax of permuted rows
    prob_measure<<<dim3(Lc, BHc), blk, 0, stream>>>(S, sample_idx + (size_t)l * Lc * Lc, Mb);
    topk_sort<<<BHc, 1024, 0, stream>>>(Mb, top);
    softmax_rows<<<dim3(Lc, BHc), blk, 0, stream>>>(S, top, P);
    // attn[z] = P[z] @ V[z]  (Bt = V^T layout), written into [B,L,H*HD] bf16
    GEMM(EPI_ATTN, P, vtB, nullptr, nullptr, nullptr, attnB, Lc, HDc, Lc, BHc,
         (long long)Lc * Lc, (long long)HDc * Lc, 0);
    // o-projection + residual, LN1
    GEMM(EPI_BIAS_RES, attnB, WoT + ow, bo + l * DMc, xf, zf, nullptr, M4c, DMc, DMc, 1, 0, 0, 0);
    ln_kernel<<<M4c, blk, 0, stream>>>(zf, g1 + l * DMc, be1 + l * DMc, xf, xb);
    // FFN
    const size_t of = (size_t)l * DMc * DFFc;
    GEMM(EPI_GELU, xb, W1T + of, b1 + l * DFFc, nullptr, nullptr, hidB, M4c, DFFc, DMc, 1, 0, 0, 0);
    GEMM(EPI_BIAS_RES, hidB, W2T + of, b2 + l * DMc, xf, zf, nullptr, M4c, DMc, DFFc, 1, 0, 0, 0);
    ln_kernel<<<M4c, blk, 0, stream>>>(zf, g2 + l * DMc, be2 + l * DMc, xf, xb);
  }

  // final LN + projection to C_OUT
  ln_kernel<<<M4c, blk, 0, stream>>>(xf, gN, bN, xf, xb);
  GEMM(EPI_BIAS, xb, WpT, b_proj, nullptr, out, nullptr, M4c, COUTc, DMc, 1, 0, 0, 0);
#undef GEMM
}
